// LSTM_Network_54022098649262
// MI455X (gfx1250) — matrix operand order repaired and checked by reading, not compiled, not run on hardware
//
#include <hip/hip_runtime.h>
#include <hip/hip_bf16.h>

typedef __attribute__((ext_vector_type(16))) __bf16 v16bf;
typedef __attribute__((ext_vector_type(8)))  float  v8f;

#define B_   128
#define T_   256
#define D_   512
#define H_   1024
#define G_   4096   /* 4*H */
#define DH_  512
#define NCLS_ 10

__device__ __forceinline__ float sigmoidf_(float x) {
    return 1.0f / (1.0f + __expf(-x));
}

// ---------------------------------------------------------------------------
// One GEMM term: 32x32 per-wave tile.
// acc[mi*2+ni] += A[m0+mi*16 : +16, 0:K] * Bt[n0+ni*16 : +16, 0:K]^T
// A  : row-major [M, lda] bf16
// Bt : row-major [N, K]  bf16 (original B transposed)
// Fragment layouts used by this code (contiguous k, same for A and B; not the ISA table order):
//   A (16x32 bf16): lanes 0-15 = rows M=0..15; lane-half h holds
//     slots 0-15 -> K = k0 + h*16 .. +15 (contiguous)
//   B (32x16 bf16): lanes 0-15 = cols N=0..15 (lanes 16-31 repeat),
//     lane-half h holds K = k0 + h*16 .. +15 (contiguous in Bt)
// Per K-step: 2 A frags + 2 B frags (8 x b128) feed 4 v_wmma -> 2.0 loads/WMMA.
// ---------------------------------------------------------------------------
__device__ __forceinline__ void gemm_term(const __bf16* __restrict__ A, int lda,
                                          const __bf16* __restrict__ Bt, int K,
                                          int m0, int n0, int lane, v8f* acc) {
    const int half = lane >> 4;     // 0 or 1
    const int r    = lane & 15;
    const __bf16* arow0 = A + (size_t)(m0 + r) * lda;
    const __bf16* arow1 = arow0 + (size_t)16 * lda;
    const __bf16* brow0 = Bt + (size_t)(n0 + r) * K + half * 16;
    const __bf16* brow1 = brow0 + (size_t)16 * K;

    for (int k0 = 0; k0 < K; k0 += 32) {
        v16bf a0, a1, b0, b1;
        a0 = *(const v16bf*)(arow0 + k0 + half * 16);
        a1 = *(const v16bf*)(arow1 + k0 + half * 16);
        b0 = *(const v16bf*)(brow0 + k0);
        b1 = *(const v16bf*)(brow1 + k0);

        acc[0] = __builtin_amdgcn_wmma_f32_16x16x32_bf16(false, a0, false, b0, (short)0, acc[0], false, false);
        acc[1] = __builtin_amdgcn_wmma_f32_16x16x32_bf16(false, a0, false, b1, (short)0, acc[1], false, false);
        acc[2] = __builtin_amdgcn_wmma_f32_16x16x32_bf16(false, a1, false, b0, (short)0, acc[2], false, false);
        acc[3] = __builtin_amdgcn_wmma_f32_16x16x32_bf16(false, a1, false, b1, (short)0, acc[3], false, false);
    }
}

// Z[M=128, N] = A0[128,K0] @ B0t^T + A1[128,K1] @ B1t^T   (raw, no bias)
// block = 256 threads (8 waves) arranged 4(M) x 2(N): covers M=128 x N=64.
// grid.x = N/64.
__global__ __launch_bounds__(256)
void wmma_gemm(const __bf16* __restrict__ A0, int lda0,
               const __bf16* __restrict__ B0t, int K0,
               const __bf16* __restrict__ A1, int lda1,
               const __bf16* __restrict__ B1t, int K1,
               float* __restrict__ Z, int N) {
    const int lane = threadIdx.x & 31;
    const int wave = threadIdx.x >> 5;
    const int m0   = (wave & 3) * 32;
    const int n0   = blockIdx.x * 64 + (wave >> 2) * 32;

    v8f acc[4] = {};
    gemm_term(A0, lda0, B0t, K0, m0, n0, lane, acc);
    if (A1 != nullptr && K1 > 0)
        gemm_term(A1, lda1, B1t, K1, m0, n0, lane, acc);

    // C/D layout: VGPR i, lanes 0-15 -> M = mt+i, lanes 16-31 -> M = mt+8+i;
    // N = nt + (lane&15)
    const int half = lane >> 4;
    const int r    = lane & 15;
#pragma unroll
    for (int mi = 0; mi < 2; ++mi) {
#pragma unroll
        for (int ni = 0; ni < 2; ++ni) {
            const v8f& a = acc[mi * 2 + ni];
            float* zp = Z + (size_t)(m0 + mi * 16 + half * 8) * N + n0 + ni * 16 + r;
#pragma unroll
            for (int i = 0; i < 8; ++i)
                zp[(size_t)i * N] = a[i];
        }
    }
}

// ---------------------------------------------------------------------------
// LSTM gate elementwise: z[B, 4H] (+bias) -> c (f32, in place), h (bf16)
// Gate order i, f, g, o at columns j, j+H, j+2H, j+3H.
// ---------------------------------------------------------------------------
__global__ __launch_bounds__(256)
void lstm_gates(const float* __restrict__ z, const float* __restrict__ bias,
                float* __restrict__ c, __bf16* __restrict__ hb) {
    int idx = blockIdx.x * blockDim.x + threadIdx.x;   // < B_*H_
    int bi  = idx >> 10;          // / H_
    int j   = idx & (H_ - 1);
    const float* zr = z + (size_t)bi * G_;

    float ig = sigmoidf_(zr[j]          + bias[j]);
    float fg = sigmoidf_(zr[j +     H_] + bias[j +     H_]);
    float gg = tanhf    (zr[j + 2 * H_] + bias[j + 2 * H_]);
    float og = sigmoidf_(zr[j + 3 * H_] + bias[j + 3 * H_]);

    float cn = fg * c[idx] + ig * gg;
    c[idx]   = cn;
    hb[idx]  = (__bf16)(og * tanhf(cn));
}

// ---------------------------------------------------------------------------
// Utility kernels
// ---------------------------------------------------------------------------
__global__ void f32_to_bf16(const float* __restrict__ in, __bf16* __restrict__ out, int n) {
    int i = blockIdx.x * blockDim.x + threadIdx.x;
    if (i < n) out[i] = (__bf16)in[i];
}

// in: [K, N] f32 row-major  ->  out: [N, K] bf16 row-major
__global__ void transpose_to_bf16(const float* __restrict__ in, __bf16* __restrict__ out,
                                  int K, int N) {
    __shared__ float tile[32][33];
    int kb = blockIdx.y * 32, nb = blockIdx.x * 32;
    int tx = threadIdx.x, ty = threadIdx.y;      // block (32, 8)
    for (int i = ty; i < 32; i += 8)
        tile[i][tx] = in[(size_t)(kb + i) * N + nb + tx];
    __syncthreads();
    for (int i = ty; i < 32; i += 8)
        out[(size_t)(nb + i) * K + kb + tx] = (__bf16)tile[tx][i];
}

__global__ void zero_b32(unsigned int* __restrict__ p, int nwords) {
    int i = blockIdx.x * blockDim.x + threadIdx.x;
    if (i < nwords) p[i] = 0u;
}

__global__ void bias_relu(const float* __restrict__ z, const float* __restrict__ bias,
                          float* __restrict__ out, int n, int ncol) {
    int i = blockIdx.x * blockDim.x + threadIdx.x;
    if (i < n) {
        float v = z[i] + bias[i % ncol];
        out[i] = v > 0.0f ? v : 0.0f;
    }
}

// One wave per batch row: logits = hid @ Wd2 + bd2, then softmax over 10 classes.
__global__ void head_softmax(const float* __restrict__ hid, const float* __restrict__ Wd2,
                             const float* __restrict__ bd2, float* __restrict__ out) {
    int row = blockIdx.x;
    int lane = threadIdx.x;      // 32 threads
    float a[NCLS_];
#pragma unroll
    for (int c = 0; c < NCLS_; ++c) a[c] = 0.0f;
    for (int k = lane; k < DH_; k += 32) {
        float h = hid[(size_t)row * DH_ + k];
#pragma unroll
        for (int c = 0; c < NCLS_; ++c)
            a[c] += h * Wd2[(size_t)k * NCLS_ + c];
    }
#pragma unroll
    for (int c = 0; c < NCLS_; ++c)
        for (int off = 16; off > 0; off >>= 1)
            a[c] += __shfl_xor(a[c], off, 32);
    if (lane == 0) {
        float m = -1e30f;
#pragma unroll
        for (int c = 0; c < NCLS_; ++c) { a[c] += bd2[c]; m = fmaxf(m, a[c]); }
        float s = 0.0f;
#pragma unroll
        for (int c = 0; c < NCLS_; ++c) { a[c] = __expf(a[c] - m); s += a[c]; }
        float inv = 1.0f / s;
#pragma unroll
        for (int c = 0; c < NCLS_; ++c) out[(size_t)row * NCLS_ + c] = a[c] * inv;
    }
}

// ---------------------------------------------------------------------------
extern "C" void kernel_launch(void* const* d_in, const int* in_sizes, int n_in,
                              void* d_out, int out_size, void* d_ws, size_t ws_size,
                              hipStream_t stream) {
    const float* x   = (const float*)d_in[0];
    const float* W1  = (const float*)d_in[1];
    const float* U1  = (const float*)d_in[2];
    const float* b1  = (const float*)d_in[3];
    const float* W2  = (const float*)d_in[4];
    const float* U2  = (const float*)d_in[5];
    const float* b2  = (const float*)d_in[6];
    const float* Wd1 = (const float*)d_in[7];
    const float* bd1 = (const float*)d_in[8];
    const float* Wd2 = (const float*)d_in[9];
    const float* bd2 = (const float*)d_in[10];
    float* out = (float*)d_out;

    // --- workspace carve-up (all regions fully rewritten every call) ---
    char* ws = (char*)d_ws;
    size_t off = 0;
    auto carve = [&](size_t bytes) {
        void* p = ws + off;
        off = (off + bytes + 255) & ~(size_t)255;
        return p;
    };
    __bf16* xb   = (__bf16*)carve((size_t)B_ * T_ * D_ * 2);   // 32 MB
    __bf16* W1t  = (__bf16*)carve((size_t)G_ * D_ * 2);        // [4096, 512]
    __bf16* U1t  = (__bf16*)carve((size_t)G_ * H_ * 2);        // [4096,1024]
    __bf16* W2t  = (__bf16*)carve((size_t)G_ * H_ * 2);
    __bf16* U2t  = (__bf16*)carve((size_t)G_ * H_ * 2);
    __bf16* Wd1t = (__bf16*)carve((size_t)DH_ * H_ * 2);       // [512, 1024]
    __bf16* h1b  = (__bf16*)carve((size_t)B_ * H_ * 2);
    __bf16* h2b  = (__bf16*)carve((size_t)B_ * H_ * 2);
    float*  c1   = (float*)carve((size_t)B_ * H_ * 4);
    float*  c2   = (float*)carve((size_t)B_ * H_ * 4);
    float*  z1   = (float*)carve((size_t)B_ * G_ * 4);         // 2 MB
    float*  z2   = (float*)carve((size_t)B_ * G_ * 4);
    float*  zh   = (float*)carve((size_t)B_ * DH_ * 4);
    float*  hid  = (float*)carve((size_t)B_ * DH_ * 4);

    // --- one-time prep: bf16 conversion + weight transposes ---
    {
        int n = B_ * T_ * D_;
        f32_to_bf16<<<n / 256, 256, 0, stream>>>(x, xb, n);
    }
    transpose_to_bf16<<<dim3(G_ / 32, D_ / 32), dim3(32, 8), 0, stream>>>(W1,  W1t,  D_, G_);
    transpose_to_bf16<<<dim3(G_ / 32, H_ / 32), dim3(32, 8), 0, stream>>>(U1,  U1t,  H_, G_);
    transpose_to_bf16<<<dim3(G_ / 32, H_ / 32), dim3(32, 8), 0, stream>>>(W2,  W2t,  H_, G_);
    transpose_to_bf16<<<dim3(G_ / 32, H_ / 32), dim3(32, 8), 0, stream>>>(U2,  U2t,  H_, G_);
    transpose_to_bf16<<<dim3(DH_ / 32, H_ / 32), dim3(32, 8), 0, stream>>>(Wd1, Wd1t, H_, DH_);

    // --- zero initial state ---
    zero_b32<<<(B_ * H_ / 2) / 256, 256, 0, stream>>>((unsigned int*)h1b, B_ * H_ / 2);
    zero_b32<<<(B_ * H_ / 2) / 256, 256, 0, stream>>>((unsigned int*)h2b, B_ * H_ / 2);
    zero_b32<<<(B_ * H_) / 256, 256, 0, stream>>>((unsigned int*)c1, B_ * H_);
    zero_b32<<<(B_ * H_) / 256, 256, 0, stream>>>((unsigned int*)c2, B_ * H_);

    // --- recurrence: 256 steps, 2 fused GEMMs + 2 gate kernels each ---
    for (int t = 0; t < T_; ++t) {
        // z1 = x_t @ W1 + h1 @ U1
        wmma_gemm<<<G_ / 64, 256, 0, stream>>>(xb + (size_t)t * D_, T_ * D_, W1t, D_,
                                               h1b, H_, U1t, H_, z1, G_);
        lstm_gates<<<(B_ * H_) / 256, 256, 0, stream>>>(z1, b1, c1, h1b);
        // z2 = h1 @ W2 + h2 @ U2
        wmma_gemm<<<G_ / 64, 256, 0, stream>>>(h1b, H_, W2t, H_,
                                               h2b, H_, U2t, H_, z2, G_);
        lstm_gates<<<(B_ * H_) / 256, 256, 0, stream>>>(z2, b2, c2, h2b);
    }

    // --- dense head ---
    wmma_gemm<<<DH_ / 64, 256, 0, stream>>>(h2b, H_, Wd1t, H_,
                                            nullptr, 0, nullptr, 0, zh, DH_);
    bias_relu<<<(B_ * DH_) / 256, 256, 0, stream>>>(zh, bd1, hid, B_ * DH_, DH_);
    head_softmax<<<B_, 32, 0, stream>>>(hid, Wd2, bd2, out);
}
